// SinkhornOTLoss_63565515981057
// MI455X (gfx1250) — matrix operand order repaired and checked by reading, not compiled, not run on hardware
//
#include <hip/hip_runtime.h>

// ---------------- problem constants ----------------
#define T_N     512
#define V_N     32000
#define ALPHA_F 20.0f
#define EPS_F   1e-16f
#define STOP_TH 0.005f
#define MAXIT   1000
#define CHK     50
#define A_MASS  (1.0f / (float)T_N)   // a = 1/T
#define B_MASS  (1.0f / (float)V_N)   // b = 1/V

// ---------------- launch geometry ----------------
#define NBLK   64                     // persistent blocks (must all be resident)
#define BLK    512                    // 16 wave32s per block
#define WPB    (BLK / 32)
#define NWAVE  (NBLK * WPB)           // 1024 waves

#define VGRP   (V_N / 16)             // 2000 row-groups for K^T matvec
#define VPAIR  (VGRP / 2)             // 1000 row-group pairs
#define TGRP   (T_N / 16)             // 32 row-groups for K matvec
#define KCH_V  (V_N / 32)             // 1000 k-chunks (K v)
#define KCH_T  (T_N / 32)             // 16 k-chunks (K^T u)

typedef __bf16 bhalf;
typedef __attribute__((ext_vector_type(16))) __bf16 v16bf;
typedef __attribute__((ext_vector_type(8)))  __bf16 v8bf;
typedef __attribute__((ext_vector_type(8)))  float  v8f;

// ---------------------------------------------------------------------------
// Device-wide barrier: generation counter with release/acquire at agent scope.
// Safe because all NBLK blocks are co-resident (persistent kernel).
// __threadfence() on every thread handles per-CU L0 invalidation in CU mode.
// ---------------------------------------------------------------------------
__device__ __forceinline__ void gsync(unsigned* cnt, unsigned* gen) {
  __threadfence();
  __syncthreads();
  if (threadIdx.x == 0) {
    unsigned g = __hip_atomic_load(gen, __ATOMIC_RELAXED, __HIP_MEMORY_SCOPE_AGENT);
    unsigned a = __hip_atomic_fetch_add(cnt, 1u, __ATOMIC_ACQ_REL, __HIP_MEMORY_SCOPE_AGENT);
    if (a == (unsigned)(NBLK - 1)) {
      __hip_atomic_store(cnt, 0u, __ATOMIC_RELAXED, __HIP_MEMORY_SCOPE_AGENT);
      __hip_atomic_fetch_add(gen, 1u, __ATOMIC_RELEASE, __HIP_MEMORY_SCOPE_AGENT);
    } else {
      while (__hip_atomic_load(gen, __ATOMIC_ACQUIRE, __HIP_MEMORY_SCOPE_AGENT) == g)
        __builtin_amdgcn_s_sleep(8);
    }
  }
  __syncthreads();
  __threadfence();
}

// ---------------------------------------------------------------------------
// K is stored pre-swizzled into the CDNA5 16-bit A-fragment layout (16Mx32K):
//   lane m     : k = 0..7  in elems 0..7,  k = 16..23 in elems 8..15
//   lane m+16  : k = 8..15 in elems 0..7,  k = 24..31 in elems 8..15
// Each lane's 16 bf16 (32B) are contiguous at tile + lane*16 -> two b128 loads.
// ---------------------------------------------------------------------------
__device__ __forceinline__ v16bf load_tile(const bhalf* __restrict__ tile, int lane) {
  const v8bf* p = (const v8bf*)(tile + lane * 16);
  v8bf lo = p[0], hi = p[1];
  v16bf a;
#pragma unroll
  for (int i = 0; i < 8; ++i) { a[i] = lo[i]; a[i + 8] = hi[i]; }
  return a;
}

// B fragment (32Kx16N, 16-bit): lane n / n+16 hold column n; element s of half h is
// k = 8h + s (s<8), 16 + 8h + (s-8) (s>=8). Input x: lane L holds vec[k=L].
// Matvec uses column 0 only; all other columns zero.
__device__ __forceinline__ v16bf vec_frag(float x) {
  v16bf b = {};
  int l = __lane_id();
#pragma unroll
  for (int s = 0; s < 16; ++s) {
    float y = __shfl(x, ((l >> 4) << 3) + ((s >> 3) << 4) + (s & 7), 32);
    if ((l & 15) == 0) b[s] = (bhalf)y;
  }
  return b;
}

__device__ __forceinline__ v8f wmma_bf16(v16bf a, v16bf b, v8f c) {
  return __builtin_amdgcn_wmma_f32_16x16x32_bf16(false, a, false, b, (short)0, c,
                                                 false, false);
}

// Two adjacent K^T row-groups (vr = 2p, 2p+1): two independent WMMA chains
// interleaved so they fill each other's hazard slots and double load ILP.
__device__ __forceinline__ void kt_pair(const bhalf* __restrict__ KTA,
                                        const float* __restrict__ u,
                                        int p, int lane, v8f& c0, v8f& c1) {
  const bhalf* tb = KTA + (size_t)p * (2 * KCH_T * 512);
  c0 = (v8f){};
  c1 = (v8f){};
#pragma unroll 2
  for (int kc = 0; kc < KCH_T; ++kc) {
    v16bf b  = vec_frag(u[kc * 32 + lane]);
    v16bf a0 = load_tile(tb + kc * 512, lane);
    v16bf a1 = load_tile(tb + (KCH_T + kc) * 512, lane);
    c0 = wmma_bf16(a0, b, c0);
    c1 = wmma_bf16(a1, b, c1);
  }
}

extern "C" __global__ void sinkhorn_init_gfx1250(unsigned* cnt, unsigned* gen, int* conv) {
  if (threadIdx.x == 0) { *cnt = 0u; *gen = 0u; *conv = 0; }
}

extern "C" __global__ void __launch_bounds__(BLK)
sinkhorn_persist_gfx1250(const float* __restrict__ cost,
                         float* __restrict__ out,
                         bhalf* __restrict__ KA, bhalf* __restrict__ KTA,
                         float* __restrict__ uvec, float* __restrict__ vvec,
                         float* __restrict__ partF, float* __restrict__ errPart,
                         float* __restrict__ bpart,
                         unsigned* cnt, unsigned* gen, int* conv)
{
  __shared__ float sred[BLK];
  const int tid  = threadIdx.x;
  const int gtid = blockIdx.x * BLK + tid;
  const int lane = tid & 31;
  const int gw   = blockIdx.x * WPB + (tid >> 5);   // global wave id, wave32

  // ---- Phase 0: build both swizzled bf16 copies of K = exp(-alpha*cost) ----
  for (int idx = gtid; idx < T_N * V_N; idx += NBLK * BLK) {
    int tile = idx >> 9, w = idx & 511;
    int ln = w >> 4, e = w & 15;
    int m = ln & 15;
    int k = ((ln >> 4) << 3) + ((e >> 3) << 4) + (e & 7);
    { // KA: tiles (tr, kc) row-group-major; rows = t, k-dim = v
      int tr = tile / KCH_V, kc = tile - tr * KCH_V;
      int t = tr * 16 + m, vv = kc * 32 + k;
      KA[idx] = (bhalf)__expf(-ALPHA_F * cost[(size_t)t * V_N + vv]);
    }
    { // KTA: tiles (vr, kc2); rows = v, k-dim = t
      int vr = tile >> 4, kc2 = tile & 15;
      int vv = vr * 16 + m, t = kc2 * 32 + k;
      KTA[idx] = (bhalf)__expf(-ALPHA_F * cost[(size_t)t * V_N + vv]);
    }
  }
  if (gtid < T_N) uvec[gtid] = A_MASS;   // u0 = a
  gsync(cnt, gen);

  // ---- Sinkhorn iterations (device-side loop + convergence break) ----
  for (int it = 0; it < MAXIT; ++it) {
    // v = b / (K^T u + eps): one wave owns a pair of 16-row groups.
    if (gw < VPAIR) {
      v8f c0, c1;
      kt_pair(KTA, uvec, gw, lane, c0, c1);
      if ((lane & 15) == 0) {            // lanes 0 / 16 hold column 0 of D
        int mb = (lane >> 4) << 3;
#pragma unroll
        for (int r = 0; r < 8; ++r) {
          vvec[(2 * gw)     * 16 + mb + r] = B_MASS * __builtin_amdgcn_rcpf(c0[r] + EPS_F);
          vvec[(2 * gw + 1) * 16 + mb + r] = B_MASS * __builtin_amdgcn_rcpf(c1[r] + EPS_F);
        }
      }
    }
    gsync(cnt, gen);

    // Partial sums of K v: 32 waves per t-row-group; each wave runs two
    // independent even/odd k-chunk chains to fill WMMA hazard slots.
    {
      int g = gw >> 5, wg = gw & 31;
      const bhalf* gb = KA + (size_t)g * KCH_V * 512;
      v8f c0 = (v8f){};
      v8f c1 = (v8f){};
      int kc = wg;
      for (; kc + 32 < KCH_V; kc += 64) {
        if (kc + 64 < KCH_V)
          __builtin_prefetch(gb + (size_t)(kc + 64) * 512 + lane * 16, 0, 1);
        v16bf a0 = load_tile(gb + (size_t)kc * 512, lane);
        v16bf b0 = vec_frag(vvec[kc * 32 + lane]);
        v16bf a1 = load_tile(gb + (size_t)(kc + 32) * 512, lane);
        v16bf b1 = vec_frag(vvec[(kc + 32) * 32 + lane]);
        c0 = wmma_bf16(a0, b0, c0);
        c1 = wmma_bf16(a1, b1, c1);
      }
      if (kc < KCH_V) {                  // odd tail chunk
        v16bf a0 = load_tile(gb + (size_t)kc * 512, lane);
        v16bf b0 = vec_frag(vvec[kc * 32 + lane]);
        c0 = wmma_bf16(a0, b0, c0);
      }
      if ((lane & 15) == 0) {
        int mb = (lane >> 4) << 3;
#pragma unroll
        for (int r = 0; r < 8; ++r)
          partF[(g * 32 + wg) * 16 + mb + r] = c0[r] + c1[r];
      }
    }
    gsync(cnt, gen);

    // u = a / (K v + eps): deterministic fixed-order combine of 32 partials.
    if (gtid < T_N) {
      int g = gtid >> 4, m = gtid & 15;
      float s = 0.f;
#pragma unroll 8
      for (int wg = 0; wg < 32; ++wg) s += partF[(g * 32 + wg) * 16 + m];
      uvec[gtid] = A_MASS * __builtin_amdgcn_rcpf(s + EPS_F);
    }
    gsync(cnt, gen);

    if (((it + 1) % CHK) == 0) {
      // err = sum |v * (K^T u_new) - b|  (matches reference check semantics)
      if (gw < VPAIR) {
        v8f c0, c1;
        kt_pair(KTA, uvec, gw, lane, c0, c1);
        if ((lane & 15) == 0) {
          int mb = (lane >> 4) << 3;
          float s0 = 0.f, s1 = 0.f;
#pragma unroll
          for (int r = 0; r < 8; ++r) {
            s0 += fabsf(vvec[(2 * gw)     * 16 + mb + r] * c0[r] - B_MASS);
            s1 += fabsf(vvec[(2 * gw + 1) * 16 + mb + r] * c1[r] - B_MASS);
          }
          errPart[(2 * gw)     * 2 + (lane >> 4)] = s0;
          errPart[(2 * gw + 1) * 2 + (lane >> 4)] = s1;
        }
      }
      gsync(cnt, gen);
      if (blockIdx.x == 0) {             // deterministic final err reduction
        float s = 0.f;
        for (int j = tid; j < 2 * VGRP; j += BLK) s += errPart[j];
        sred[tid] = s; __syncthreads();
        for (int st = BLK / 2; st > 0; st >>= 1) {
          if (tid < st) sred[tid] += sred[tid + st];
          __syncthreads();
        }
        if (tid == 0) *conv = (sred[0] < STOP_TH) ? 1 : 0;
      }
      gsync(cnt, gen);
      if (__hip_atomic_load(conv, __ATOMIC_RELAXED, __HIP_MEMORY_SCOPE_AGENT))
        break;                           // uniform across the whole grid
    }
  }

  // ---- loss = 100 * sum(u_t * K * v_v * cost), K recomputed on the fly ----
  float part = 0.f;
  for (int t = blockIdx.x; t < T_N; t += NBLK) {
    const float* crow = cost + (size_t)t * V_N;
    float rs = 0.f;
    for (int j = tid; j < V_N; j += BLK) {
      float cc = crow[j];
      rs += __expf(-ALPHA_F * cc) * vvec[j] * cc;
    }
    part += uvec[t] * rs;
  }
  sred[tid] = part; __syncthreads();
  for (int st = BLK / 2; st > 0; st >>= 1) {
    if (tid < st) sred[tid] += sred[tid + st];
    __syncthreads();
  }
  if (tid == 0) bpart[blockIdx.x] = sred[0];
  gsync(cnt, gen);
  if (blockIdx.x == 0) {
    float s = (tid < NBLK) ? bpart[tid] : 0.f;
    sred[tid] = s; __syncthreads();
    for (int st = BLK / 2; st > 0; st >>= 1) {
      if (tid < st) sred[tid] += sred[tid + st];
      __syncthreads();
    }
    if (tid == 0) out[0] = 100.0f * sred[0];
  }
}

// ---------------------------------------------------------------------------
extern "C" void kernel_launch(void* const* d_in, const int* in_sizes, int n_in,
                              void* d_out, int out_size, void* d_ws, size_t ws_size,
                              hipStream_t stream) {
  (void)in_sizes; (void)n_in; (void)out_size;
  const float* cost = (const float*)d_in[0];
  float* out = (float*)d_out;

  char* ws = (char*)d_ws;
  size_t off = 0;
  auto take = [&](size_t bytes) -> void* {
    void* p = ws + off;
    off += (bytes + 255) & ~(size_t)255;
    return p;
  };
  bhalf*    KA    = (bhalf*)take((size_t)T_N * V_N * sizeof(bhalf));   // 32.77 MB
  bhalf*    KTA   = (bhalf*)take((size_t)T_N * V_N * sizeof(bhalf));   // 32.77 MB
  float*    u     = (float*)take((size_t)T_N * sizeof(float));
  float*    v     = (float*)take((size_t)V_N * sizeof(float));
  float*    partF = (float*)take((size_t)TGRP * 32 * 16 * sizeof(float));
  float*    errP  = (float*)take((size_t)2 * VGRP * sizeof(float));
  float*    bpart = (float*)take((size_t)NBLK * sizeof(float));
  unsigned* cnt   = (unsigned*)take(sizeof(unsigned));
  unsigned* gen   = (unsigned*)take(sizeof(unsigned));
  int*      conv  = (int*)take(sizeof(int));

  if (off > ws_size) return;  // workspace too small: refuse to corrupt memory

  // Barrier/flag state must be zeroed every call (ws is poisoned, not restored).
  sinkhorn_init_gfx1250<<<1, 64, 0, stream>>>(cnt, gen, conv);
  sinkhorn_persist_gfx1250<<<NBLK, BLK, 0, stream>>>(
      cost, out, KA, KTA, u, v, partF, errP, bpart, cnt, gen, conv);
}
